// GatherConvND_1571958030441
// MI455X (gfx1250) — matrix operand order repaired and checked by reading, not compiled, not run on hardware
//
#include <hip/hip_runtime.h>
#include <hip/hip_bf16.h>
#include <cstddef>

// ---------------------------------------------------------------------------
// GatherConvND for MI455X (gfx1250, wave32).
// Two 4096x1024x1024 GEMMs run on v_wmma_f32_16x16x32_bf16 with LDS staging,
// 128x128 block tiles, 8 waves, async global->LDS copies (ASYNCcnt-tracked).
// ---------------------------------------------------------------------------

typedef __attribute__((ext_vector_type(16))) __bf16        v16bf;
typedef __attribute__((ext_vector_type(8)))  float         v8f;
typedef __attribute__((ext_vector_type(4)))  unsigned int  u32x4;
typedef __attribute__((ext_vector_type(4)))  float         f32x4;
typedef __attribute__((ext_vector_type(4)))  unsigned short u16x4;

// Exact vector type expected by the async-LDS builtin (gcc-style v4i32).
typedef int gvec4i __attribute__((vector_size(16)));
#define AS1 __attribute__((address_space(1)))
#define AS3 __attribute__((address_space(3)))

#define L_SEQ 4096
#define C_DIM 1024
#define N_DIM 1024
#define H_HEADS 16
#define D_HEAD 64
#define K_PROF 64
#define S_SAMP 33
#define MAX_FREQ 16.0f
#define MIN_FREQ 1.0f
#define MAX_RECEPTIVE 256.0f

// gfx1250 async global->LDS (ASYNCcnt-tracked) if this toolchain exposes it.
#if __has_builtin(__builtin_amdgcn_global_load_async_to_lds_b128) && \
    __has_builtin(__builtin_amdgcn_s_wait_asynccnt)
#define USE_ASYNC_LDS 1
#else
#define USE_ASYNC_LDS 0
#endif

__device__ __forceinline__ float sigmoid_f(float v) { return 1.0f / (1.0f + __expf(-v)); }
__device__ __forceinline__ float silu_f(float v)    { return v * sigmoid_f(v); }

__device__ __forceinline__ unsigned short f32_to_bf16_rne(float f) {
    unsigned int u = __float_as_uint(f);
    unsigned int rounding = 0x7FFFu + ((u >> 16) & 1u);
    return (unsigned short)((u + rounding) >> 16);
}

// ---------------------------------------------------------------------------
// fp32 -> bf16 conversion (n multiple of 4)
// ---------------------------------------------------------------------------
__global__ void cvt_f32_to_bf16_kernel(const float* __restrict__ in,
                                       unsigned short* __restrict__ out, int n) {
    int i = (blockIdx.x * blockDim.x + threadIdx.x) * 4;
    if (i >= n) return;
    f32x4 v = *(const f32x4*)(in + i);
    u16x4 o;
    o.x = f32_to_bf16_rne(v.x);
    o.y = f32_to_bf16_rne(v.y);
    o.z = f32_to_bf16_rne(v.z);
    o.w = f32_to_bf16_rne(v.w);
    *(u16x4*)(out + i) = o;
}

// ---------------------------------------------------------------------------
// Wave projection: one wave (32 lanes) per sequence position.
// Lane j computes output j of silu(x @ Ww^T + bw); lanes 0..15 -> freq,
// lanes 16..31 -> phase; head-mean reduced through LDS.
// ---------------------------------------------------------------------------
__global__ __launch_bounds__(32) void wave_proj_kernel(
    const float* __restrict__ x, const float* __restrict__ Ww,
    const float* __restrict__ bw, float* __restrict__ freq_avg,
    float* __restrict__ phase_avg) {
    const int l = blockIdx.x;
    const int lane = threadIdx.x;
    __shared__ float row[C_DIM];
    __shared__ float red[32];

    const float* xr = x + (size_t)l * C_DIM;
    for (int c = lane * 4; c < C_DIM; c += 32 * 4)
        *(f32x4*)(row + c) = *(const f32x4*)(xr + c);
    __syncthreads();

    const float* w = Ww + (size_t)lane * C_DIM;
    float acc = bw[lane];
    for (int c = 0; c < C_DIM; c += 4) {
        acc += row[c] * w[c] + row[c + 1] * w[c + 1] +
               row[c + 2] * w[c + 2] + row[c + 3] * w[c + 3];
    }
    float wp = silu_f(acc);
    float val = (lane < 16) ? (sigmoid_f(wp) * (MAX_FREQ - MIN_FREQ) + MIN_FREQ)
                            : (tanhf(wp) * MAX_FREQ);
    red[lane] = val;
    __syncthreads();
    if (lane == 0) {
        float fs = 0.f, ps = 0.f;
        for (int h = 0; h < 16; ++h) { fs += red[h]; ps += red[16 + h]; }
        freq_avg[l]  = fs * (1.0f / 16.0f);
        phase_avg[l] = ps * (1.0f / 16.0f);
    }
}

// ---------------------------------------------------------------------------
// Tiled bf16 WMMA GEMM: out[m,n] = silu(sum_k A[m,k]*Bw[n,k] + bias[n]).
// A: MxK bf16 row-major. Bw: NxK bf16 row-major (out = A * Bw^T).
// BM=BN=128, BK=32, 256 threads = 8 waves; wave (r,c) with r=wave>>1,
// c=wave&1 computes rows [r*32, r*32+32) x cols [c*64, c*64+64):
// 2 A-frags x 4 B-frags = 8 v_wmma_f32_16x16x32_bf16 per K-step.
// ---------------------------------------------------------------------------
#define BM 128
#define BN 128
#define BK 32
#define GEMM_THREADS 256
#define LDS_STRIDE 40  // bf16 elements; 80B row keeps 16B alignment, skews banks

union FragAB { u32x4 q[2]; v16bf v; };

__global__ __launch_bounds__(GEMM_THREADS) void gemm_bf16_silu_kernel(
    const unsigned short* __restrict__ A,
    const unsigned short* __restrict__ Bw,
    const float* __restrict__ bias,          // may be nullptr
    float* __restrict__ outF,                // may be nullptr
    unsigned short* __restrict__ outB,       // may be nullptr
    int M, int N, int Kc) {
    __shared__ unsigned short As[BM][LDS_STRIDE];
    __shared__ unsigned short Bs[BN][LDS_STRIDE];

    const int tid  = threadIdx.x;
    const int wave = tid >> 5;
    const int lane = tid & 31;
    const int n0 = blockIdx.x * BN;
    const int m0 = blockIdx.y * BM;
    const int wm = (wave >> 1) * 32;   // wave's M offset within block tile
    const int wn = (wave & 1) * 64;    // wave's N offset within block tile

    v8f acc[2][4] = {};

    for (int k0 = 0; k0 < Kc; k0 += BK) {
        // Stage 128x32 bf16 tiles of A and Bw into LDS: 512 16B chunks each,
        // 2 per thread per matrix. Branch-free; prefetch is speculative so
        // running past the last tile is harmless.
#pragma unroll
        for (int rep = 0; rep < 2; ++rep) {
            int i = tid + rep * GEMM_THREADS;  // 0..511
            int row = i >> 2;
            int quad = i & 3;
            const unsigned short* srcA = A  + (size_t)(m0 + row) * Kc + k0 + quad * 8;
            const unsigned short* srcB = Bw + (size_t)(n0 + row) * Kc + k0 + quad * 8;
#if USE_ASYNC_LDS
            __builtin_amdgcn_global_load_async_to_lds_b128(
                (AS1 gvec4i*)srcA, (AS3 gvec4i*)&As[row][quad * 8], 0, 0);
            __builtin_amdgcn_global_load_async_to_lds_b128(
                (AS1 gvec4i*)srcB, (AS3 gvec4i*)&Bs[row][quad * 8], 0, 0);
#else
            *(u32x4*)&As[row][quad * 8] = *(const u32x4*)srcA;
            *(u32x4*)&Bs[row][quad * 8] = *(const u32x4*)srcB;
#endif
            __builtin_prefetch((const void*)(srcA + BK), 0, 1);
            __builtin_prefetch((const void*)(srcB + BK), 0, 1);
        }
#if USE_ASYNC_LDS
        __builtin_amdgcn_s_wait_asynccnt(0);
#endif
        __syncthreads();

        // A fragments (16x32 bf16 each): lane<16 holds M=lane&15, K 0..7 &
        // 16..23; lane>=16 holds same M, K 8..15 & 24..31.
        FragAB a[2];
#pragma unroll
        for (int ai = 0; ai < 2; ++ai) {
            int m  = wm + ai * 16 + (lane & 15);
            int cb = (lane >> 4) * 8;
            a[ai].q[0] = *(const u32x4*)&As[m][cb];
            a[ai].q[1] = *(const u32x4*)&As[m][cb + 16];
        }
        // B fragments (32x16 bf16): lane column = lane&15; lane<16 holds
        // K 0..7 & 16..23; lane>=16 holds K 8..15 & 24..31.
#pragma unroll
        for (int j = 0; j < 4; ++j) {
            FragAB b;
            int n  = wn + j * 16 + (lane & 15);
            int kb = (lane >> 4) * 8;
            b.q[0] = *(const u32x4*)&Bs[n][kb];
            b.q[1] = *(const u32x4*)&Bs[n][kb + 16];
#pragma unroll
            for (int ai = 0; ai < 2; ++ai) {
                acc[ai][j] = __builtin_amdgcn_wmma_f32_16x16x32_bf16(
                    false, a[ai].v, false, b.v, (short)0, acc[ai][j], false, false);
            }
        }
        __syncthreads();
    }

    // Epilogue: C/D layout — VGPR r: lane<16 -> (M=r, N=lane),
    // lane>=16 -> (M=r+8, N=lane-16).
    const int rbase = (lane >> 4) * 8;
    const int col0  = lane & 15;
#pragma unroll
    for (int ai = 0; ai < 2; ++ai) {
#pragma unroll
        for (int j = 0; j < 4; ++j) {
            int col = n0 + wn + j * 16 + col0;
            float bv = bias ? bias[col] : 0.0f;
#pragma unroll
            for (int r = 0; r < 8; ++r) {
                int row = m0 + wm + ai * 16 + rbase + r;
                float v = acc[ai][j][r] + bv;
                v = silu_f(v);
                size_t off = (size_t)row * N + col;
                if (outF) outF[off] = v;
                if (outB) outB[off] = f32_to_bf16_rne(v);
            }
        }
    }
}

// ---------------------------------------------------------------------------
// Gather + kernel interpolation + weighted sum. One block per position l,
// 16 waves = one per head; lane covers 2 of the 64 head channels.
// ---------------------------------------------------------------------------
__global__ __launch_bounds__(512) void gather_kernel(
    const float* __restrict__ x, const float* __restrict__ kmax,
    const float* __restrict__ freq_avg, const float* __restrict__ phase_avg,
    unsigned short* __restrict__ midb) {
    const int l = blockIdx.x;
    const int tid = threadIdx.x;
    const int h = tid >> 5;
    const int lane = tid & 31;

    __shared__ int   s_idx[S_SAMP];
    __shared__ float s_w[H_HEADS][S_SAMP];
    __shared__ float s_inv[H_HEADS];
    __shared__ float s_fa, s_pa;

    if (tid == 0) { s_fa = freq_avg[l]; s_pa = phase_avg[l]; }
    __syncthreads();
    const float fa = s_fa, pa = s_pa;

    // Per-head sample weights (lane s; lane 0 also covers s=32).
    for (int s = lane; s < S_SAMP; s += 32) {
        float rel = (float)(s - 16) * fa + pa;
        float sp  = (float)l + rel;
        bool  valid = (sp >= 0.0f) && (sp < (float)L_SEQ);
        int idx = (int)sp;                        // trunc toward zero (.long())
        idx = idx < 0 ? 0 : (idx > L_SEQ - 1 ? L_SEQ - 1 : idx);
        if (h == 0) s_idx[s] = idx;

        float np = (rel + MAX_RECEPTIVE) * (1.0f / (2.0f * MAX_RECEPTIVE));
        np = np < 0.0f ? 0.0f : (np > 1.0f ? 1.0f : np);
        float fi  = np * (float)(K_PROF - 1);
        int   ifl = (int)fi;
        ifl = ifl < 0 ? 0 : (ifl > K_PROF - 2 ? K_PROF - 2 : ifl);
        float wc  = fi - (float)ifl;
        const float* kp = kmax + (size_t)l * (H_HEADS * K_PROF) + h * K_PROF;
        float kf = kp[ifl], kc = kp[ifl + 1];
        s_w[h][s] = (kf * (1.0f - wc) + kc * wc) * (valid ? 1.0f : 0.0f);
    }
    __syncthreads();

    if (lane == 0) {
        float sum = 0.0f;
        for (int s = 0; s < S_SAMP; ++s) sum += s_w[h][s];
        s_inv[h] = 1.0f / (sum + 1e-8f);
    }
    __syncthreads();

    const float inv = s_inv[h];
    const int dbase = h * D_HEAD;
    float acc0 = 0.0f, acc1 = 0.0f;
    for (int s = 0; s < S_SAMP; ++s) {
        float w = s_w[h][s] * inv;
        const float* xp = x + (size_t)s_idx[s] * C_DIM + dbase;
        acc0 += w * xp[lane];
        acc1 += w * xp[lane + 32];
    }
    unsigned short* mp = midb + (size_t)l * C_DIM + dbase;
    mp[lane]      = f32_to_bf16_rne(acc0);
    mp[lane + 32] = f32_to_bf16_rne(acc1);
}

// ---------------------------------------------------------------------------
// Host-side orchestration
// ---------------------------------------------------------------------------
extern "C" void kernel_launch(void* const* d_in, const int* in_sizes, int n_in,
                              void* d_out, int out_size, void* d_ws, size_t ws_size,
                              hipStream_t stream) {
    const float* x  = (const float*)d_in[0];
    const float* Ww = (const float*)d_in[1];
    const float* bw = (const float*)d_in[2];
    const float* Wk = (const float*)d_in[3];
    const float* bk = (const float*)d_in[4];
    const float* Wo = (const float*)d_in[5];
    float* out = (float*)d_out;

    const int L = L_SEQ, C = C_DIM, N = N_DIM;

    // Workspace carve-up (all offsets multiples of 256B).
    char* ws = (char*)d_ws;
    unsigned short* x_b   = (unsigned short*)ws; ws += (size_t)L * C * 2;  // 8 MB
    unsigned short* wk_b  = (unsigned short*)ws; ws += (size_t)N * C * 2;  // 2 MB
    unsigned short* wo_b  = (unsigned short*)ws; ws += (size_t)N * C * 2;  // 2 MB
    unsigned short* mid_b = (unsigned short*)ws; ws += (size_t)L * C * 2;  // 8 MB
    float* kmax = (float*)ws; ws += (size_t)L * N * 4;                     // 16 MB
    float* fa   = (float*)ws; ws += (size_t)L * 4;
    float* pa   = (float*)ws; ws += (size_t)L * 4;

    // 1) fp32 -> bf16 conversions.
    {
        int n = L * C;
        cvt_f32_to_bf16_kernel<<<(n / 4 + 255) / 256, 256, 0, stream>>>(x, x_b, n);
        n = N * C;
        cvt_f32_to_bf16_kernel<<<(n / 4 + 255) / 256, 256, 0, stream>>>(Wk, wk_b, n);
        cvt_f32_to_bf16_kernel<<<(n / 4 + 255) / 256, 256, 0, stream>>>(Wo, wo_b, n);
    }

    // 2) Wave projection -> freq_avg / phase_avg.
    wave_proj_kernel<<<L, 32, 0, stream>>>(x, Ww, bw, fa, pa);

    // 3) kernel_max = silu(x @ Wk^T + bk)  [WMMA bf16 GEMM, fp32 out].
    {
        dim3 grid(N / BN, L / BM);
        gemm_bf16_silu_kernel<<<grid, GEMM_THREADS, 0, stream>>>(
            x_b, wk_b, bk, kmax, (unsigned short*)nullptr, L, N, C);
    }

    // 4) Gather + interpolate + normalize + weighted sum -> mid (bf16).
    gather_kernel<<<L, 512, 0, stream>>>(x, kmax, fa, pa, mid_b);

    // 5) out = silu(mid @ Wo^T)  [WMMA bf16 GEMM, fp32 out].
    {
        dim3 grid(N / BN, L / BM);
        gemm_bf16_silu_kernel<<<grid, GEMM_THREADS, 0, stream>>>(
            mid_b, wo_b, (const float*)nullptr, out, (unsigned short*)nullptr, L, N, C);
    }
}
